// FCN_DAttn_71073118814786
// MI455X (gfx1250) — matrix operand order repaired and checked by reading, not compiled, not run on hardware
//
#include <hip/hip_runtime.h>
#include <cstdint>
#include <math.h>

// ---------------------------------------------------------------------------
// Types for CDNA5 WMMA
// ---------------------------------------------------------------------------
typedef __attribute__((ext_vector_type(16))) _Float16 v16h;
typedef __attribute__((ext_vector_type(8)))  float    v8f;

#define NSPAT 4104     // 513*8 spatial positions for all post-conv1 tensors
#define BATCH 4

// ---------------------------------------------------------------------------
// f32 -> f16 weight conversion
// ---------------------------------------------------------------------------
__global__ void cvt_f16_kernel(const float* __restrict__ in, _Float16* __restrict__ out, int n) {
    int i = blockIdx.x * blockDim.x + threadIdx.x;
    if (i < n) out[i] = (_Float16)in[i];
}

// ---------------------------------------------------------------------------
// conv1: fuses the unfold/reshape of x into the 8x8 stride-4 conv.
// c_in[b,0,p,q] (p<4096,q<61): lin=p*61+q, w=lin>>12, s=lin&4095, j=(w<<11)+s,
// value = (j<126975) ? x[b,j] : 0   (the single padded zero).
// Output h1 [B,128,1025,16] f16 (ReLU applied).
// ---------------------------------------------------------------------------
__global__ void conv1_kernel(const float* __restrict__ x, const float* __restrict__ w1,
                             const float* __restrict__ b1, _Float16* __restrict__ h1) {
    int idx = blockIdx.x * blockDim.x + threadIdx.x;
    const int total = BATCH * 128 * 1025 * 16;
    if (idx >= total) return;
    int ow = idx & 15;
    int t  = idx >> 4;
    int oh = t % 1025; t /= 1025;
    int oc = t & 127;
    int b  = t >> 7;
    const float* wr = w1 + oc * 64;
    const float* xb = x + (size_t)b * 126975;
    float sum = b1[oc];
#pragma unroll
    for (int kh = 0; kh < 8; ++kh) {
        int p = 4 * oh - 4 + kh;
        if ((unsigned)p >= 4096u) continue;
#pragma unroll
        for (int kw = 0; kw < 8; ++kw) {
            int q = 4 * ow - 4 + kw;
            if ((unsigned)q >= 61u) continue;
            int lin = p * 61 + q;
            int w = lin >> 12;
            int s = lin & 4095;
            int j = (w << 11) + s;
            float xv = (j < 126975) ? xb[j] : 0.0f;
            sum += wr[kh * 8 + kw] * xv;
        }
    }
    h1[idx] = (_Float16)fmaxf(sum, 0.0f);
}

// ---------------------------------------------------------------------------
// Implicit-GEMM conv with WMMA f32<-f16, all geometry compile-time.
//   D[m,n] = sum_k A[m,k]*B[k,n]; A = weights [OC][K] f16 (OIHW flat),
//   B[k,n] = im2col of act [IC][IH][IW] f16.
// One wave computes MT stacked 16(M) x 16(N) tiles; the B fragment is
// gathered ONCE per 32-wide k-step and reused by MT WMMAs.
// B gather is BRANCHLESS: every element loads from a clamped-safe address and
// selects zero afterwards -> 16 independent u16 loads (clause-able, single
// wait) instead of 16 exec-predicated serialized loads.
// A fragment (16-bit A 16x32): lane l holds M=l&15;
//   VGPR v: k = (v&3)*2 + 8*(l>=16) + 16*(v>=4)  (pair).
// B fragment (16-bit B 32x16): lane l holds N=l&15;
//   VGPR v: k = (v&3)*2 + 8*(l>=16) + 16*(v>=4) (pair), same as A.
// ---------------------------------------------------------------------------
template <int OC, int IC, int IH, int IW, int KH, int KW, int STRIDE, int PAD, int OW, int K,
          int MT>
__global__ __launch_bounds__(32) void wmma_conv_kernel(
    const _Float16* __restrict__ A, const _Float16* __restrict__ act,
    float* __restrict__ out32, _Float16* __restrict__ out16,
    const float* __restrict__ bias,
    const float* __restrict__ bns, const float* __restrict__ bnb,
    const float* __restrict__ bnm, const float* __restrict__ bnv,
    int N) {
    const int lane = threadIdx.x;
    const int col  = lane & 15;
    const int hi   = lane >> 4;
    const int n0   = blockIdx.x * 16;
    const int m0   = blockIdx.y * (16 * MT);
    const int b    = blockIdx.z;

    const int n = n0 + col;
    const bool nval = (n < N);
    const int oh = n / OW;                    // OW==8 -> shift
    const int ow = n % OW;                    // -> mask
    const int ih_base = oh * STRIDE - PAD;
    const int iw_base = ow * STRIDE - PAD;

    const uint16_t* actb = reinterpret_cast<const uint16_t*>(act) + (size_t)b * IC * IH * IW;
    const _Float16* Arow = A + (size_t)(m0 + col) * K;
    const int aKoff = hi * 8;
    const int bKoff = hi * 8;

    v8f acc[MT];
#pragma unroll
    for (int mt = 0; mt < MT; ++mt) acc[mt] = {};

    for (int k0 = 0; k0 < K; k0 += 32) {
        // ---- B fragment: 16 k's in two runs of 8 (same order as A), branchless clamped gather ----
        uint32_t bv[16];
        {
            const int kstart = k0 + bKoff;
            int kw = kstart % KW;              // compile-time divisor
            int tt = kstart / KW;
            int kh = tt % KH;
            int ic = tt / KH;
#pragma unroll
            for (int t = 0; t < 16; ++t) {
                int ih = ih_base + kh;
                int iw = iw_base + kw;
                bool inb = nval & ((unsigned)ih < (unsigned)IH) & ((unsigned)iw < (unsigned)IW);
                size_t off = inb ? (((size_t)ic * IH + ih) * IW + iw) : (size_t)0;
                uint32_t raw = actb[off];      // always-safe address
                bv[t] = inb ? raw : 0u;
                {
                    int kn = kstart + t + 1 + ((t + 1) >= 8 ? 8 : 0);
                    kw = kn % KW; int tn = kn / KW; kh = tn % KH; ic = tn / KH;
                }
            }
        }
        v16h bfrag;
        uint32_t* bu = reinterpret_cast<uint32_t*>(&bfrag);
#pragma unroll
        for (int vv = 0; vv < 8; ++vv) bu[vv] = (bv[2 * vv] & 0xffffu) | (bv[2 * vv + 1] << 16);

        // ---- MT x (A fragment load + WMMA), sharing bfrag ----
#pragma unroll
        for (int mt = 0; mt < MT; ++mt) {
            v16h afrag;
            uint32_t* ap = reinterpret_cast<uint32_t*>(&afrag);
            const _Float16* Ar = Arow + (size_t)mt * 16 * K;
#pragma unroll
            for (int vv = 0; vv < 8; ++vv) {
                int kk = k0 + ((vv & 3) * 2) + aKoff + ((vv >> 2) * 16);
                ap[vv] = *reinterpret_cast<const uint32_t*>(Ar + kk);
            }
            acc[mt] = __builtin_amdgcn_wmma_f32_16x16x32_f16(
                false, afrag, false, bfrag, (short)0, acc[mt], false, false);
        }
    }

    if (!nval) return;
#pragma unroll
    for (int mt = 0; mt < MT; ++mt) {
#pragma unroll
        for (int r = 0; r < 8; ++r) {
            int m = m0 + mt * 16 + r + hi * 8;
            float val = acc[mt][r];
            if (bias) val += bias[m];
            if (bns) {
                float inv = bns[m] * rsqrtf(bnv[m] + 1e-5f);
                val = (val - bnm[m]) * inv + bnb[m];
            }
            val = fmaxf(val, 0.0f);
            size_t o = ((size_t)b * OC + m) * N + n;
            if (out32) out32[o] = val;
            if (out16) out16[o] = (_Float16)val;
        }
    }
}

// ---------------------------------------------------------------------------
// PAM q/k/v 1x1 convs (f32 VALU; tiny)
// ---------------------------------------------------------------------------
__global__ void pam_qkv_kernel(const float* __restrict__ feat1,
                               const float* __restrict__ qw, const float* __restrict__ qb,
                               const float* __restrict__ kw, const float* __restrict__ kb,
                               const float* __restrict__ vw, const float* __restrict__ vb,
                               float* __restrict__ q, float* __restrict__ k,
                               float* __restrict__ v, int N) {
    int n = blockIdx.x * blockDim.x + threadIdx.x;
    int b = blockIdx.y;
    if (n >= N) return;
    const float* f1 = feat1 + (size_t)b * 32 * N;
    float f[32];
#pragma unroll
    for (int c = 0; c < 32; ++c) f[c] = f1[c * N + n];
#pragma unroll
    for (int o = 0; o < 4; ++o) {
        float sq = qb[o], sk = kb[o];
#pragma unroll
        for (int c = 0; c < 32; ++c) { sq += qw[o * 32 + c] * f[c]; sk += kw[o * 32 + c] * f[c]; }
        q[((size_t)b * 4 + o) * N + n] = sq;
        k[((size_t)b * 4 + o) * N + n] = sk;
    }
    for (int o = 0; o < 32; ++o) {
        float sv = vb[o];
#pragma unroll
        for (int c = 0; c < 32; ++c) sv += vw[o * 32 + c] * f[c];
        v[((size_t)b * 32 + o) * N + n] = sv;
    }
}

// ---------------------------------------------------------------------------
// PAM flash attention: out[b,c,i] = sum_j softmax_j(q_i . k_j) v[c,j]
// Single pass, online softmax, 32 f32 accumulators per thread.
// Never materializes the 4x4104x4104 (269 MB) attention matrix.
// Epilogue: sa = gamma*out + feat1 -> f16 (input to conv51).
// ---------------------------------------------------------------------------
__global__ void pam_flash_kernel(const float* __restrict__ q, const float* __restrict__ k,
                                 const float* __restrict__ v, const float* __restrict__ feat1,
                                 const float* __restrict__ gamma,
                                 _Float16* __restrict__ sa16, int N) {
    int i = blockIdx.x * blockDim.x + threadIdx.x;
    int b = blockIdx.y;
    if (i >= N) return;
    const float* qb = q + (size_t)b * 4 * N;
    const float* kb = k + (size_t)b * 4 * N;
    const float* vb = v + (size_t)b * 32 * N;
    float q0 = qb[i], q1 = qb[N + i], q2 = qb[2 * N + i], q3 = qb[3 * N + i];
    float m = -INFINITY, l = 0.0f;
    float acc[32];
#pragma unroll
    for (int c = 0; c < 32; ++c) acc[c] = 0.0f;
    for (int j = 0; j < N; ++j) {
        float s = q0 * kb[j] + q1 * kb[N + j] + q2 * kb[2 * N + j] + q3 * kb[3 * N + j];
        float nm = fmaxf(m, s);
        float corr = __expf(m - nm);   // first iter: exp(-inf)=0
        float p = __expf(s - nm);
        l = l * corr + p;
#pragma unroll
        for (int c = 0; c < 32; ++c) acc[c] = acc[c] * corr + p * vb[(size_t)c * N + j];
        m = nm;
    }
    float g = gamma[0];
    float invl = 1.0f / l;
    const float* f1 = feat1 + (size_t)b * 32 * N;
#pragma unroll
    for (int c = 0; c < 32; ++c) {
        float o = g * (acc[c] * invl) + f1[c * N + i];
        sa16[((size_t)b * 32 + c) * N + i] = (_Float16)o;
    }
}

// ---------------------------------------------------------------------------
// CAM: energy = Xf Xf^T (32x32), cattn = softmax(rowmax - energy)
//   softmax(max - e)_d == exp(-e_d)/sum exp(-e_d'); use row-min for stability.
// ---------------------------------------------------------------------------
__global__ __launch_bounds__(1024) void cam_energy_kernel(const float* __restrict__ feat2,
                                                          float* __restrict__ cattn, int N) {
    __shared__ float e[32 * 32];
    int b = blockIdx.x;
    int c = threadIdx.x >> 5;
    int d = threadIdx.x & 31;
    const float* fb = feat2 + (size_t)b * 32 * N;
    float s = 0.0f;
    for (int n = 0; n < N; ++n) s += fb[c * N + n] * fb[d * N + n];
    e[threadIdx.x] = s;
    __syncthreads();
    if (threadIdx.x < 32) {
        int row = threadIdx.x;
        float mn = e[row * 32];
        for (int j = 1; j < 32; ++j) mn = fminf(mn, e[row * 32 + j]);
        float p[32];
        float sum = 0.0f;
        for (int j = 0; j < 32; ++j) { p[j] = __expf(mn - e[row * 32 + j]); sum += p[j]; }
        float inv = 1.0f / sum;
        for (int j = 0; j < 32; ++j) cattn[((size_t)b * 32 + row) * 32 + j] = p[j] * inv;
    }
}

__global__ __launch_bounds__(256) void cam_apply_kernel(const float* __restrict__ feat2,
                                                        const float* __restrict__ cattn,
                                                        const float* __restrict__ gamma,
                                                        _Float16* __restrict__ sc16, int N) {
    __shared__ float cat[1024];
    int b = blockIdx.y;
    for (int t = threadIdx.x; t < 1024; t += 256) cat[t] = cattn[(size_t)b * 1024 + t];
    __syncthreads();
    int n = blockIdx.x * 256 + threadIdx.x;
    if (n >= N) return;
    const float* f2 = feat2 + (size_t)b * 32 * N;
    float f[32];
#pragma unroll
    for (int d = 0; d < 32; ++d) f[d] = f2[d * N + n];
    float g = gamma[0];
#pragma unroll
    for (int c = 0; c < 32; ++c) {
        float s = 0.0f;
#pragma unroll
        for (int d = 0; d < 32; ++d) s += cat[c * 32 + d] * f[d];
        sc16[((size_t)b * 32 + c) * N + n] = (_Float16)(g * s + f[c]);
    }
}

// ---------------------------------------------------------------------------
// Head: out[b,0,cls,oh,ow] = c8_b[cls] + sum_c c8_w[cls,c]*(sa_conv+sc_conv)
// ---------------------------------------------------------------------------
__global__ void head_kernel(const float* __restrict__ saconv, const float* __restrict__ scconv,
                            const float* __restrict__ w8, const float* __restrict__ b8,
                            float* __restrict__ out, int N) {
    int n = blockIdx.x * blockDim.x + threadIdx.x;
    int b = blockIdx.y;
    if (n >= N) return;
    const float* sa = saconv + (size_t)b * 32 * N;
    const float* sc = scconv + (size_t)b * 32 * N;
    float s0 = b8[0], s1 = b8[1];
#pragma unroll
    for (int c = 0; c < 32; ++c) {
        float x = sa[c * N + n] + sc[c * N + n];
        s0 += w8[c] * x;
        s1 += w8[32 + c] * x;
    }
    out[((size_t)b * 2 + 0) * N + n] = s0;
    out[((size_t)b * 2 + 1) * N + n] = s1;
}

// ---------------------------------------------------------------------------
// Host orchestration
// ---------------------------------------------------------------------------
extern "C" void kernel_launch(void* const* d_in, const int* in_sizes, int n_in,
                              void* d_out, int out_size, void* d_ws, size_t ws_size,
                              hipStream_t stream) {
    (void)in_sizes; (void)n_in; (void)out_size; (void)ws_size;
    const int N = NSPAT;

    const float* x      = (const float*)d_in[0];
    const float* w1     = (const float*)d_in[1];
    const float* b1     = (const float*)d_in[2];
    const float* w2     = (const float*)d_in[3];
    const float* b2     = (const float*)d_in[4];
    const float* w3     = (const float*)d_in[5];
    const float* b3     = (const float*)d_in[6];
    const float* c5a_w  = (const float*)d_in[7];
    const float* bn5a_s = (const float*)d_in[8];
    const float* bn5a_b = (const float*)d_in[9];
    const float* bn5a_m = (const float*)d_in[10];
    const float* bn5a_v = (const float*)d_in[11];
    const float* c5c_w  = (const float*)d_in[12];
    const float* bn5c_s = (const float*)d_in[13];
    const float* bn5c_b = (const float*)d_in[14];
    const float* bn5c_m = (const float*)d_in[15];
    const float* bn5c_v = (const float*)d_in[16];
    const float* c51_w  = (const float*)d_in[17];
    const float* bn51_s = (const float*)d_in[18];
    const float* bn51_b = (const float*)d_in[19];
    const float* bn51_m = (const float*)d_in[20];
    const float* bn51_v = (const float*)d_in[21];
    const float* c52_w  = (const float*)d_in[22];
    const float* bn52_s = (const float*)d_in[23];
    const float* bn52_b = (const float*)d_in[24];
    const float* bn52_m = (const float*)d_in[25];
    const float* bn52_v = (const float*)d_in[26];
    const float* pq_w   = (const float*)d_in[27];
    const float* pq_b   = (const float*)d_in[28];
    const float* pk_w   = (const float*)d_in[29];
    const float* pk_b   = (const float*)d_in[30];
    const float* pv_w   = (const float*)d_in[31];
    const float* pv_b   = (const float*)d_in[32];
    const float* pgamma = (const float*)d_in[33];
    const float* cgamma = (const float*)d_in[34];
    const float* c8_w   = (const float*)d_in[35];
    const float* c8_b   = (const float*)d_in[36];
    float* out = (float*)d_out;

    // ---- workspace bump allocator (256B aligned) ----
    uint8_t* wsb = (uint8_t*)d_ws;
    size_t off = 0;
    auto alloc = [&](size_t bytes) -> void* {
        void* p = wsb + off;
        off += (bytes + 255) & ~(size_t)255;
        return p;
    };
    _Float16* w2h   = (_Float16*)alloc((size_t)819200 * 2);
    _Float16* w3h   = (_Float16*)alloc((size_t)294912 * 2);
    _Float16* w5ah  = (_Float16*)alloc((size_t)36864 * 2);
    _Float16* w5ch  = (_Float16*)alloc((size_t)36864 * 2);
    _Float16* w51h  = (_Float16*)alloc((size_t)9216 * 2);
    _Float16* w52h  = (_Float16*)alloc((size_t)9216 * 2);
    _Float16* h1    = (_Float16*)alloc((size_t)BATCH * 128 * 1025 * 16 * 2);
    _Float16* h2    = (_Float16*)alloc((size_t)BATCH * 256 * N * 2);
    _Float16* h3    = (_Float16*)alloc((size_t)BATCH * 128 * N * 2);
    float*    feat1 = (float*)alloc((size_t)BATCH * 32 * N * 4);
    float*    feat2 = (float*)alloc((size_t)BATCH * 32 * N * 4);
    float*    qbuf  = (float*)alloc((size_t)BATCH * 4 * N * 4);
    float*    kbuf  = (float*)alloc((size_t)BATCH * 4 * N * 4);
    float*    vbuf  = (float*)alloc((size_t)BATCH * 32 * N * 4);
    _Float16* sa16  = (_Float16*)alloc((size_t)BATCH * 32 * N * 2);
    _Float16* sc16  = (_Float16*)alloc((size_t)BATCH * 32 * N * 2);
    float*    cattn = (float*)alloc((size_t)BATCH * 32 * 32 * 4);
    float*    saconv= (float*)alloc((size_t)BATCH * 32 * N * 4);
    float*    scconv= (float*)alloc((size_t)BATCH * 32 * N * 4);

    // ---- weight conversions to f16 ----
    auto cvt = [&](const float* src, _Float16* dst, int n) {
        cvt_f16_kernel<<<(n + 255) / 256, 256, 0, stream>>>(src, dst, n);
    };
    cvt(w2, w2h, 819200);
    cvt(w3, w3h, 294912);
    cvt(c5a_w, w5ah, 36864);
    cvt(c5c_w, w5ch, 36864);
    cvt(c51_w, w51h, 9216);
    cvt(c52_w, w52h, 9216);

    // ---- conv1 (fused unfold), f32 -> f16 ----
    {
        int total = BATCH * 128 * 1025 * 16;
        conv1_kernel<<<(total + 255) / 256, 256, 0, stream>>>(x, w1, b1, h1);
    }

    const int ntiles = (N + 15) / 16;   // 257

    // ---- conv2: 256x4104x3200 WMMA (MT=4 -> 64 M rows/wave), bias+ReLU -> f16
    wmma_conv_kernel<256, 128, 1025, 16, 5, 5, 2, 2, 8, 3200, 4>
        <<<dim3(ntiles, 256 / 64, BATCH), 32, 0, stream>>>(
            w2h, h1, nullptr, h2, b2, nullptr, nullptr, nullptr, nullptr, N);

    // ---- conv3: 128x4104x2304 WMMA (MT=4), bias+ReLU -> f16 ----
    wmma_conv_kernel<128, 256, 513, 8, 3, 3, 1, 1, 8, 2304, 4>
        <<<dim3(ntiles, 128 / 64, BATCH), 32, 0, stream>>>(
            w3h, h2, nullptr, h3, b3, nullptr, nullptr, nullptr, nullptr, N);

    // ---- conv5a / conv5c: 32x4104x1152 WMMA (MT=2), BN+ReLU -> f32 ----
    wmma_conv_kernel<32, 128, 513, 8, 3, 3, 1, 1, 8, 1152, 2>
        <<<dim3(ntiles, 1, BATCH), 32, 0, stream>>>(
            w5ah, h3, feat1, nullptr, nullptr, bn5a_s, bn5a_b, bn5a_m, bn5a_v, N);
    wmma_conv_kernel<32, 128, 513, 8, 3, 3, 1, 1, 8, 1152, 2>
        <<<dim3(ntiles, 1, BATCH), 32, 0, stream>>>(
            w5ch, h3, feat2, nullptr, nullptr, bn5c_s, bn5c_b, bn5c_m, bn5c_v, N);

    // ---- PAM ----
    pam_qkv_kernel<<<dim3((N + 127) / 128, BATCH), 128, 0, stream>>>(
        feat1, pq_w, pq_b, pk_w, pk_b, pv_w, pv_b, qbuf, kbuf, vbuf, N);
    pam_flash_kernel<<<dim3((N + 127) / 128, BATCH), 128, 0, stream>>>(
        qbuf, kbuf, vbuf, feat1, pgamma, sa16, N);

    // ---- CAM ----
    cam_energy_kernel<<<BATCH, 1024, 0, stream>>>(feat2, cattn, N);
    cam_apply_kernel<<<dim3((N + 255) / 256, BATCH), 256, 0, stream>>>(
        feat2, cattn, cgamma, sc16, N);

    // ---- conv51 / conv52: 32x4104x288 WMMA (MT=2), BN+ReLU -> f32 ----
    wmma_conv_kernel<32, 32, 513, 8, 3, 3, 1, 1, 8, 288, 2>
        <<<dim3(ntiles, 1, BATCH), 32, 0, stream>>>(
            w51h, sa16, saconv, nullptr, nullptr, bn51_s, bn51_b, bn51_m, bn51_v, N);
    wmma_conv_kernel<32, 32, 513, 8, 3, 3, 1, 1, 8, 288, 2>
        <<<dim3(ntiles, 1, BATCH), 32, 0, stream>>>(
            w52h, sc16, scconv, nullptr, nullptr, bn52_s, bn52_b, bn52_m, bn52_v, N);

    // ---- head (conv8) ----
    head_kernel<<<dim3((N + 255) / 256, BATCH), 256, 0, stream>>>(
        saconv, scconv, c8_w, c8_b, out, N);
}
